// Long2DSCSelfAttention_54451595378934
// MI455X (gfx1250) — matrix operand order repaired and checked by reading, not compiled, not run on hardware
//
#include <hip/hip_runtime.h>
#include <hip/hip_bf16.h>

typedef __attribute__((ext_vector_type(16))) __bf16 v16bf;
typedef __attribute__((ext_vector_type(8)))  __bf16 v8bf;
typedef __attribute__((ext_vector_type(8)))  float  v8f;

#define NEGC (-1.0e9f)

__device__ __forceinline__ v8f wmma_bf16(v16bf a, v16bf b, v8f c){
  return __builtin_amdgcn_wmma_f32_16x16x32_bf16(false, a, false, b, (short)0, c, false, false);
}

__device__ __forceinline__ v16bf join8(v8bf lo, v8bf hi){
  return __builtin_shufflevector(lo, hi, 0,1,2,3,4,5,6,7,8,9,10,11,12,13,14,15);
}

// A fragment from row-major [16 x stride] bf16 tile: two 16B spans per lane.
__device__ __forceinline__ v16bf load_a_frag(const __bf16* base, int stride, int lane){
  const int r = lane & 15, hi = lane >> 4;
  const __bf16* p = base + r * stride + hi * 8;
  return join8(*(const v8bf*)p, *(const v8bf*)(p + 16));
}

// B fragment from k-contiguous [N][Kstride] layout: two 16B spans per lane.
__device__ __forceinline__ v16bf load_bk_frag(const __bf16* base, int stride, int lane){
  const int n = lane & 15, hi = lane >> 4;
  const __bf16* p = base + n * stride + hi * 8;
  return join8(*(const v8bf*)p, *(const v8bf*)(p + 16));
}

__device__ __forceinline__ v8bf zero8(){
  v8bf z;
#pragma unroll
  for (int i = 0; i < 8; ++i) z[i] = (__bf16)0.f;
  return z;
}

// CDNA5 async copy: 16 bytes per lane, global -> LDS, tracked by ASYNCcnt.
__device__ __forceinline__ void async_copy16(__bf16* lds, const __bf16* g){
  const unsigned lo = (unsigned)(unsigned long long)lds;
  asm volatile("global_load_async_to_lds_b128 %0, %1, off" :: "v"(lo), "v"(g) : "memory");
}
__device__ __forceinline__ void wait_async(){
  asm volatile("s_wait_asynccnt 0" ::: "memory");
}

__device__ __forceinline__ void store_out(float* p, float v){ *p = v; }
__device__ __forceinline__ void store_out(__bf16* p, float v){ *p = (__bf16)v; }

// ---------------------------------------------------------------------------
__global__ void convert_f32_bf16(const float* __restrict__ in, __bf16* __restrict__ out, int n4){
  const int i = blockIdx.x * 256 + threadIdx.x;
  if (i < n4){
    const float4 v = ((const float4*)in)[i];
    union { uint2 u; __bf16 h[4]; } z;
    z.h[0] = (__bf16)v.x; z.h[1] = (__bf16)v.y; z.h[2] = (__bf16)v.z; z.h[3] = (__bf16)v.w;
    ((uint2*)out)[i] = z.u;
  }
}

__global__ void transpose_f32_bf16(const float* __restrict__ w, __bf16* __restrict__ wT, int K, int N){
  const int idx = blockIdx.x * 256 + threadIdx.x;
  if (idx < K * N){
    const int k = idx / N, n = idx - k * N;
    wT[(long)n * K + k] = (__bf16)w[idx];
  }
}

// ---------------------------------------------------------------------------
// Batched GEMM: C[b] = alpha * A[b](MxK,bf16) @ W (WT=[N][K] bf16) (+ bias)
// ---------------------------------------------------------------------------
template <typename OutT>
__global__ __launch_bounds__(256)
void gemm_bf16_wmma(const __bf16* __restrict__ A, const __bf16* __restrict__ WT,
                    const float* __restrict__ bias, OutT* __restrict__ C,
                    int M, int N, int K, long strideA, long strideC,
                    int lda, int ldc, float alpha)
{
  __shared__ alignas(16) __bf16 sA[128 * 32];   // [row][k]
  __shared__ alignas(16) __bf16 sBt[128 * 32];  // [col][k]

  const int tid  = threadIdx.x;
  const int lane = tid & 31;
  const int wave = tid >> 5;
  const int wr   = wave >> 1;
  const int wc   = wave & 1;
  const int rowBase = blockIdx.y * 128;
  const int colBase = blockIdx.x * 128;
  const int b = blockIdx.z;
  const bool fullA = (rowBase + 128 <= M);

  const __bf16* Ab = A + (long)b * strideA;
  OutT* Cb = C + (long)b * strideC;

  v8f acc[8];
#pragma unroll
  for (int i = 0; i < 8; ++i) acc[i] = (v8f){0.f,0.f,0.f,0.f,0.f,0.f,0.f,0.f};

  for (int k0 = 0; k0 < K; k0 += 32){
    if (fullA){
#pragma unroll
      for (int u = 0; u < 2; ++u){
        const int ve = tid + u * 256;
        const int r = ve >> 2, c8 = (ve & 3) << 3;
        async_copy16(sA + r * 32 + c8, Ab + (long)(rowBase + r) * lda + k0 + c8);
      }
    } else {
      for (int ve = tid; ve < 128 * 4; ve += 256){
        const int r = ve >> 2, c8 = (ve & 3) << 3;
        const int gr = rowBase + r;
        v8bf v = zero8();
        if (gr < M) v = *(const v8bf*)(Ab + (long)gr * lda + k0 + c8);
        *(v8bf*)(sA + r * 32 + c8) = v;
      }
    }
#pragma unroll
    for (int u = 0; u < 2; ++u){
      const int ve = tid + u * 256;
      const int nn = ve >> 2, c8 = (ve & 3) << 3;
      async_copy16(sBt + nn * 32 + c8, WT + (long)(colBase + nn) * K + k0 + c8);
    }
    wait_async();
    __syncthreads();

    v16bf a0 = load_a_frag(sA + (wr * 32)      * 32, 32, lane);
    v16bf a1 = load_a_frag(sA + (wr * 32 + 16) * 32, 32, lane);
    v16bf b0 = load_bk_frag(sBt + (wc * 64)      * 32, 32, lane);
    v16bf b1 = load_bk_frag(sBt + (wc * 64 + 16) * 32, 32, lane);
    v16bf b2 = load_bk_frag(sBt + (wc * 64 + 32) * 32, 32, lane);
    v16bf b3 = load_bk_frag(sBt + (wc * 64 + 48) * 32, 32, lane);
    acc[0] = wmma_bf16(a0, b0, acc[0]);
    acc[1] = wmma_bf16(a0, b1, acc[1]);
    acc[2] = wmma_bf16(a0, b2, acc[2]);
    acc[3] = wmma_bf16(a0, b3, acc[3]);
    acc[4] = wmma_bf16(a1, b0, acc[4]);
    acc[5] = wmma_bf16(a1, b1, acc[5]);
    acc[6] = wmma_bf16(a1, b2, acc[6]);
    acc[7] = wmma_bf16(a1, b3, acc[7]);
    __syncthreads();
  }

  const int n  = lane & 15, hi = lane >> 4;
#pragma unroll
  for (int r2 = 0; r2 < 2; ++r2){
#pragma unroll
    for (int c4 = 0; c4 < 4; ++c4){
      const v8f& a = acc[r2 * 4 + c4];
      const int gc = colBase + wc * 64 + c4 * 16 + n;
      const float bv = bias ? bias[gc] : 0.f;
#pragma unroll
      for (int r = 0; r < 8; ++r){
        const int row = rowBase + wr * 32 + r2 * 16 + r + (hi << 3);
        if (row < M) store_out(&Cb[(long)row * ldc + gc], a[r] * alpha + bv);
      }
    }
  }
}

// ---------------------------------------------------------------------------
__global__ __launch_bounds__(256)
void local_window_attn(const __bf16* __restrict__ qbuf,
                       const __bf16* __restrict__ kvbuf,
                       const float* __restrict__ lbt,
                       const float* __restrict__ g2l,
                       const int*   __restrict__ rel,
                       __bf16* __restrict__ x1pre)
{
  __shared__ alignas(16) __bf16 sQ [64 * 32];
  __shared__ alignas(16) __bf16 sK [64 * 32];
  __shared__ alignas(16) __bf16 sVt[32 * 64];
  __shared__ alignas(16) __bf16 sP [64 * 64];
  __shared__ float  sS[64 * 68];
  __shared__ float  sO[64 * 32];
  __shared__ float  sMr[64], sLr[64], sAl[64];
  __shared__ float  sKg[32], sVg[32];

  const int tid  = threadIdx.x;
  const int lane = tid & 31;
  const int wave = tid >> 5;
  const int wi = blockIdx.x >> 3, wj = blockIdx.x & 7;
  const int h = blockIdx.y;
  const int b = blockIdx.z;
  const long kvB = (long)b * 3137 * 768;

  for (int ve = tid; ve < 64 * 4; ve += 256){
    const int t = ve >> 2, c8 = (ve & 3) << 3;
    if (t >= 49){
      *(v8bf*)(sQ + t * 32 + c8) = zero8();
      *(v8bf*)(sK + t * 32 + c8) = zero8();
    }
  }
  for (int ve = tid; ve < 49 * 4; ve += 256){
    const int t = ve >> 2, c8 = (ve & 3) << 3;
    const int a = t / 7, bb = t - a * 7;
    const int p = (wi * 7 + a) * 56 + (wj * 7 + bb);
    async_copy16(sQ + t * 32 + c8, qbuf + ((long)b * 3136 + p) * 384 + h * 32 + c8);
  }
  if (tid < 32){
    sKg[tid] = (float)kvbuf[kvB + h * 32 + tid];
    sVg[tid] = (float)kvbuf[kvB + 384 + h * 32 + tid];
  }
  wait_async();
  __syncthreads();

  if (tid < 64){
    float m0 = 0.f;
    if (tid < 49){
      float s = 0.f;
#pragma unroll
      for (int m = 0; m < 32; ++m) s += (float)sQ[tid * 32 + m] * sKg[m];
      s += g2l[1 * 12 + h];
      m0 = s;
    }
    sMr[tid] = m0; sLr[tid] = 1.f;
  }
  for (int e = tid; e < 64 * 32; e += 256){
    const int t = e >> 5, m = e & 31;
    sO[e] = (t < 49) ? sVg[m] : 0.f;
  }
  __syncthreads();

  for (int c = 0; c < 9; ++c){
    const int ni = wi + c / 3 - 1;
    const int nj = wj + c % 3 - 1;
    if (ni < 0 || ni >= 8 || nj < 0 || nj >= 8) continue;

    for (int ve = tid; ve < 49 * 4; ve += 256){
      const int t = ve >> 2, c8 = (ve & 3) << 3;
      const int a = t / 7, bb = t - a * 7;
      const long row = (long)b * 3137 + 1 + (ni * 7 + a) * 56 + (nj * 7 + bb);
      async_copy16(sK + t * 32 + c8, kvbuf + row * 768 + h * 32 + c8);
    }
    for (int pe = tid; pe < 32 * 32; pe += 256){
      const int tp = pe >> 5;
      const int m  = pe & 31;
      const int t0 = tp * 2, t1 = t0 + 1;
      union { unsigned u; __bf16 h2[2]; } z;
      const int a0 = t0 / 7, b0 = t0 - a0 * 7;
      const int a1 = t1 / 7, b1 = t1 - a1 * 7;
      z.h2[0] = (t0 < 49) ? kvbuf[((long)b * 3137 + 1 + (ni * 7 + a0) * 56 + (nj * 7 + b0)) * 768 + 384 + h * 32 + m] : (__bf16)0.f;
      z.h2[1] = (t1 < 49) ? kvbuf[((long)b * 3137 + 1 + (ni * 7 + a1) * 56 + (nj * 7 + b1)) * 768 + 384 + h * 32 + m] : (__bf16)0.f;
      *(unsigned*)(sVt + m * 64 + t0) = z.u;
    }
    wait_async();
    __syncthreads();

#pragma unroll
    for (int td = 0; td < 2; ++td){
      const int tile = wave * 2 + td;
      const int ti = tile >> 2, tj = tile & 3;
      v16bf a  = load_a_frag (sQ + ti * 16 * 32, 32, lane);
      v16bf bt = load_bk_frag(sK + tj * 16 * 32, 32, lane);
      v8f z = {0.f,0.f,0.f,0.f,0.f,0.f,0.f,0.f};
      v8f s = wmma_bf16(a, bt, z);
      const int n = lane & 15, hi = lane >> 4;
#pragma unroll
      for (int r = 0; r < 8; ++r)
        sS[(ti * 16 + r + (hi << 3)) * 68 + tj * 16 + n] = s[r];
    }
    __syncthreads();

    if (tid < 64){
      const int r = tid;
      const float mOld = sMr[r];
      float rmax = NEGC;
      if (r < 49){
        const int* ri = rel + r * 441 + c * 49;
        for (int t = 0; t < 49; ++t){
          const float s = sS[r * 68 + t] + lbt[ri[t] * 12 + h];
          sS[r * 68 + t] = s;
          rmax = fmaxf(rmax, s);
        }
      }
      float mNew = fmaxf(mOld, rmax);
      float al   = __expf(mOld - mNew);
      float sum  = 0.f;
      if (r >= 49){ mNew = 0.f; al = 1.f; }
      for (int t2 = 0; t2 < 32; ++t2){
        const int t0 = t2 * 2, t1 = t0 + 1;
        float p0 = 0.f, p1 = 0.f;
        if (r < 49){
          if (t0 < 49){ p0 = __expf(sS[r * 68 + t0] - mNew); sum += p0; }
          if (t1 < 49){ p1 = __expf(sS[r * 68 + t1] - mNew); sum += p1; }
        }
        union { unsigned u; __bf16 h2[2]; } z;
        z.h2[0] = (__bf16)p0; z.h2[1] = (__bf16)p1;
        *(unsigned*)(sP + r * 64 + t0) = z.u;
      }
      sMr[r] = mNew; sLr[r] = sLr[r] * al + sum; sAl[r] = al;
    }
    __syncthreads();

    for (int e = tid; e < 64 * 32; e += 256) sO[e] *= sAl[e >> 5];
    __syncthreads();

    {
      const int ti = wave >> 1, tj = wave & 1;
      v16bf a0 = load_a_frag (sP + ti * 16 * 64,      64, lane);
      v16bf a1 = load_a_frag (sP + ti * 16 * 64 + 32, 64, lane);
      v16bf b0 = load_bk_frag(sVt + (tj * 16) * 64,      64, lane);
      v16bf b1 = load_bk_frag(sVt + (tj * 16) * 64 + 32, 64, lane);
      v8f acc = {0.f,0.f,0.f,0.f,0.f,0.f,0.f,0.f};
      acc = wmma_bf16(a0, b0, acc);
      acc = wmma_bf16(a1, b1, acc);
      const int n = lane & 15, hi = lane >> 4;
#pragma unroll
      for (int r = 0; r < 8; ++r)
        sO[(ti * 16 + r + (hi << 3)) * 32 + tj * 16 + n] += acc[r];
    }
    __syncthreads();
  }

  for (int e = tid; e < 49 * 32; e += 256){
    const int t = e >> 5, m = e & 31;
    const int a = t / 7, bb = t - a * 7;
    const int p = (wi * 7 + a) * 56 + (wj * 7 + bb);
    x1pre[((long)b * 3136 + p) * 384 + h * 32 + m] = (__bf16)(sO[t * 32 + m] / sLr[t]);
  }
}

// ---------------------------------------------------------------------------
__global__ __launch_bounds__(256)
void global_attn(const __bf16* __restrict__ qgbuf,
                 const float* __restrict__ kvgbuf,
                 const float* __restrict__ g2l,
                 const float* __restrict__ g2g,
                 __bf16* __restrict__ x0buf)
{
  __shared__ float sS[3137];
  __shared__ float sred[256];
  __shared__ float qv[32];
  const int tid = threadIdx.x;
  const int h = blockIdx.x, b = blockIdx.y;
  const long base = (long)b * 3137 * 768;

  if (tid < 32) qv[tid] = (float)qgbuf[b * 384 + h * 32 + tid];
  __syncthreads();

  float lmax = NEGC;
  for (int n = tid; n < 3137; n += 256){
    const float* kr = kvgbuf + base + (long)n * 768 + h * 32;
    float s = 0.f;
#pragma unroll
    for (int m = 0; m < 32; ++m) s += qv[m] * kr[m];
    s += (n == 0) ? g2g[h] : g2l[0 * 12 + h];
    sS[n] = s;
    lmax = fmaxf(lmax, s);
  }
  sred[tid] = lmax; __syncthreads();
  for (int off = 128; off > 0; off >>= 1){
    if (tid < off) sred[tid] = fmaxf(sred[tid], sred[tid + off]);
    __syncthreads();
  }
  const float mx = sred[0];
  __syncthreads();

  float ls = 0.f;
  for (int n = tid; n < 3137; n += 256){
    const float p = __expf(sS[n] - mx);
    sS[n] = p; ls += p;
  }
  sred[tid] = ls; __syncthreads();
  for (int off = 128; off > 0; off >>= 1){
    if (tid < off) sred[tid] += sred[tid + off];
    __syncthreads();
  }
  const float L = sred[0];
  __syncthreads();

  const int m = tid & 31, g = tid >> 5;
  float acc = 0.f;
  for (int n = g; n < 3137; n += 8)
    acc += sS[n] * kvgbuf[base + (long)n * 768 + 384 + h * 32 + m];
  sred[g * 32 + m] = acc; __syncthreads();
  if (tid < 32){
    float t = 0.f;
#pragma unroll
    for (int gg = 0; gg < 8; ++gg) t += sred[gg * 32 + tid];
    x0buf[b * 384 + h * 32 + tid] = (__bf16)(t / L);
  }
}

// ---------------------------------------------------------------------------
extern "C" void kernel_launch(void* const* d_in, const int* in_sizes, int n_in,
                              void* d_out, int out_size, void* d_ws, size_t ws_size,
                              hipStream_t stream)
{
  const float* x      = (const float*)d_in[0];
  const float* wq     = (const float*)d_in[1];
  const float* wkv    = (const float*)d_in[2];
  const float* wproj  = (const float*)d_in[3];
  const float* bproj  = (const float*)d_in[4];
  const float* wqg    = (const float*)d_in[5];
  const float* wkvg   = (const float*)d_in[6];
  const float* wprojg = (const float*)d_in[7];
  const float* bprojg = (const float*)d_in[8];
  const float* lbt    = (const float*)d_in[9];
  const float* g2l    = (const float*)d_in[10];
  const float* g2g    = (const float*)d_in[11];
  const int*   rel    = (const int*)d_in[12];

  const int B = 4, N = 3137, C = 384, Himg = 3136;
  const float scale = 0.17677669529663687f; // 32^-0.5

  char* ws = (char*)d_ws;
  __bf16* xbf     = (__bf16*)ws; ws += (size_t)B * N * C * 2;
  __bf16* wqT     = (__bf16*)ws; ws += (size_t)C * C * 2;
  __bf16* wkvT    = (__bf16*)ws; ws += (size_t)C * 2 * C * 2;
  __bf16* wkvgT   = (__bf16*)ws; ws += (size_t)C * 2 * C * 2;
  __bf16* wprojT  = (__bf16*)ws; ws += (size_t)C * C * 2;
  __bf16* wprojgT = (__bf16*)ws; ws += (size_t)C * C * 2;
  __bf16* wqgT    = (__bf16*)ws; ws += (size_t)C * C * 2;
  __bf16* qbuf    = (__bf16*)ws; ws += (size_t)B * Himg * C * 2;
  __bf16* kvbuf   = (__bf16*)ws; ws += (size_t)B * N * 2 * C * 2;
  float*  kvgbuf  = (float*)ws;  ws += (size_t)B * N * 2 * C * 4;
  __bf16* qgbuf   = (__bf16*)ws; ws += (size_t)B * C * 2;
  __bf16* x1pre   = (__bf16*)ws; ws += (size_t)B * Himg * C * 2;
  __bf16* x0buf   = (__bf16*)ws; ws += (size_t)B * C * 2;

  float* out = (float*)d_out;
  dim3 blk(256);

  const int xn4 = B * N * C / 4;
  convert_f32_bf16<<<dim3((xn4 + 255) / 256), blk, 0, stream>>>(x, xbf, xn4);
  transpose_f32_bf16<<<dim3((C * C + 255) / 256),     blk, 0, stream>>>(wq,     wqT,     C, C);
  transpose_f32_bf16<<<dim3((C * 2 * C + 255) / 256), blk, 0, stream>>>(wkv,    wkvT,    C, 2 * C);
  transpose_f32_bf16<<<dim3((C * 2 * C + 255) / 256), blk, 0, stream>>>(wkvg,   wkvgT,   C, 2 * C);
  transpose_f32_bf16<<<dim3((C * C + 255) / 256),     blk, 0, stream>>>(wproj,  wprojT,  C, C);
  transpose_f32_bf16<<<dim3((C * C + 255) / 256),     blk, 0, stream>>>(wprojg, wprojgT, C, C);
  transpose_f32_bf16<<<dim3((C * C + 255) / 256),     blk, 0, stream>>>(wqg,    wqgT,    C, C);

  gemm_bf16_wmma<__bf16><<<dim3(C / 128, (Himg + 127) / 128, B), blk, 0, stream>>>(
      xbf + C, wqT, nullptr, qbuf, Himg, C, C, (long)N * C, (long)Himg * C, C, C, scale);
  gemm_bf16_wmma<__bf16><<<dim3(2 * C / 128, (N + 127) / 128, B), blk, 0, stream>>>(
      xbf, wkvT, nullptr, kvbuf, N, 2 * C, C, (long)N * C, (long)N * 2 * C, C, 2 * C, 1.f);
  gemm_bf16_wmma<float><<<dim3(2 * C / 128, (N + 127) / 128, B), blk, 0, stream>>>(
      xbf, wkvgT, nullptr, kvgbuf, N, 2 * C, C, (long)N * C, (long)N * 2 * C, C, 2 * C, 1.f);
  gemm_bf16_wmma<__bf16><<<dim3(C / 128, 1, B), blk, 0, stream>>>(
      xbf, wqgT, nullptr, qgbuf, 1, C, C, (long)N * C, (long)C, C, C, scale);
  local_window_attn<<<dim3(64, 12, B), blk, 0, stream>>>(qbuf, kvbuf, lbt, g2l, rel, x1pre);
  gemm_bf16_wmma<float><<<dim3(C / 128, (Himg + 127) / 128, B), blk, 0, stream>>>(
      x1pre, wprojT, bproj, out + C, Himg, C, C, (long)Himg * C, (long)N * C, C, C, 1.f);
  global_attn<<<dim3(12, B), blk, 0, stream>>>(qgbuf, kvgbuf, g2l, g2g, x0buf);
  gemm_bf16_wmma<float><<<dim3(C / 128, 1, B), blk, 0, stream>>>(
      x0buf, wprojgT, bprojg, out, 1, C, C, (long)C, (long)N * C, C, C, 1.f);
}
